// CrossAttentionEncoder_66571993088195
// MI455X (gfx1250) — matrix operand order repaired and checked by reading, not compiled, not run on hardware
//
#include <hip/hip_runtime.h>
#include <hip/hip_bf16.h>

typedef _Float16 half_t;
typedef __attribute__((ext_vector_type(16))) _Float16 v16h;
typedef __attribute__((ext_vector_type(8)))  float    v8f;
typedef __attribute__((ext_vector_type(4)))  unsigned u32x4;
typedef __attribute__((ext_vector_type(8)))  int      i32x8;
typedef __attribute__((ext_vector_type(4)))  int      i32x4;

#define EPSV 1e-5f

#if __has_builtin(__builtin_amdgcn_tensor_load_to_lds) && \
    __has_builtin(__builtin_amdgcn_s_wait_tensorcnt)
#define USE_TDM 1
#else
#define USE_TDM 0
#endif

// ---------------------------------------------------------------------------
// WMMA helpers (CDNA5 v_wmma_f32_16x16x32_f16, wave32 layouts per ISA 7.12.2)
// ---------------------------------------------------------------------------
__device__ __forceinline__ v8f wmma32(v16h a, v16h b, v8f c) {
  return __builtin_amdgcn_wmma_f32_16x16x32_f16(
      /*neg_a=*/false, a, /*neg_b=*/false, b,
      /*c_mod=*/(short)0, c, /*reuse_a=*/false, /*reuse_b=*/false);
}

// A fragment: 16x32 f16 tile from row-major f16 matrix (row stride lda halves).
__device__ __forceinline__ v16h load_a_f16(const half_t* __restrict__ W,
                                           int lda, int r0, int k0, int lane) {
  int r  = r0 + (lane & 15);
  int kb = k0 + ((lane >> 4) << 3);
  const half_t* p0 = W + (size_t)r * lda + kb;
  const half_t* p1 = p0 + 16;
  v16h a;
#pragma unroll
  for (int e = 0; e < 8; ++e) { a[e] = p0[e]; a[e + 8] = p1[e]; }
  return a;
}

// B fragment: 32x16 (KxN) from LDS laid out [n][k] (k contiguous, stride ldb).
__device__ __forceinline__ v16h load_b_lds(const half_t* S, int ldb,
                                           int n0, int k0, int lane) {
  const half_t* p = S + (size_t)(n0 + (lane & 15)) * ldb + k0 + ((lane >> 4) << 3);
  v16h b;
#pragma unroll
  for (int e = 0; e < 8; ++e) { b[e] = p[e]; b[e + 8] = p[e + 16]; }
  return b;
}

// ---------------------------------------------------------------------------
// Prep kernels
// ---------------------------------------------------------------------------
__global__ void fuse_qkv_kernel(const float* __restrict__ Wk, const float* __restrict__ bk,
                                const float* __restrict__ Wq, const float* __restrict__ bq,
                                const float* __restrict__ Wv, const float* __restrict__ bv,
                                const float* __restrict__ W_lso, const float* __restrict__ b_lso,
                                const float* __restrict__ W_lsq, const float* __restrict__ b_lsq,
                                half_t* __restrict__ Wkh, float* __restrict__ bkp,
                                half_t* __restrict__ Wqh, float* __restrict__ bqp,
                                half_t* __restrict__ Wvh, float* __restrict__ bvp) {
  int mat = blockIdx.x >> 8;       // 0=key, 1=query, 2=value
  int o   = blockIdx.x & 255;      // output channel
  const float* W  = (mat == 0) ? Wk : ((mat == 1) ? Wq : Wv);
  const float* bs = (mat == 0) ? bk : ((mat == 1) ? bq : bv);
  const float* L  = (mat == 1) ? W_lsq : W_lso;
  const float* bl = (mat == 1) ? b_lsq : b_lso;
  half_t* Wout = (mat == 0) ? Wkh : ((mat == 1) ? Wqh : Wvh);
  float*  bout = (mat == 0) ? bkp : ((mat == 1) ? bqp : bvp);

  __shared__ float srow[256];
  int t = threadIdx.x;             // 128 threads
  srow[t] = W[o * 256 + t];
  srow[t + 128] = W[o * 256 + t + 128];
  __syncthreads();

  float acc = 0.f;
  for (int d = 0; d < 256; ++d) acc += srow[d] * L[d * 128 + t];
  Wout[o * 128 + t] = (half_t)acc;
  if (t == 0) {
    float bb = bs[o];
    for (int d = 0; d < 256; ++d) bb += srow[d] * bl[d];
    bout[o] = bb;
  }
}

__global__ void f16_convert_kernel(half_t* __restrict__ dst,
                                   const float* __restrict__ src, int n) {
  int i = blockIdx.x * 256 + threadIdx.x;
  if (i < n) dst[i] = (half_t)src[i];
}

__global__ void bn_fold_kernel(float* __restrict__ sc, float* __restrict__ sh,
                               const float* __restrict__ g, const float* __restrict__ bt,
                               const float* __restrict__ m, const float* __restrict__ v,
                               int n) {
  int i = blockIdx.x * blockDim.x + threadIdx.x;
  if (i < n) {
    float s = g[i] * rsqrtf(v[i] + EPSV);
    sc[i] = s;
    sh[i] = bt[i] - m[i] * s;
  }
}

// ---------------------------------------------------------------------------
// KNN: per-thread 16-deep register insertion sort, LDS-tiled candidate scan.
// ---------------------------------------------------------------------------
__global__ __launch_bounds__(256)
void knn_kernel(const float* __restrict__ pos, int* __restrict__ idx) {
  int b = blockIdx.x >> 4;
  int n = ((blockIdx.x & 15) << 8) + threadIdx.x;
  const float* P = pos + (size_t)b * 3 * 4096;
  float qx = P[n], qy = P[4096 + n], qz = P[8192 + n];

  float bd[16]; int bi[16];
#pragma unroll
  for (int j = 0; j < 16; ++j) { bd[j] = 3.4e38f; bi[j] = 0; }

  __shared__ float sp[3][256];
  for (int m0 = 0; m0 < 4096; m0 += 256) {
    __syncthreads();
    sp[0][threadIdx.x] = P[m0 + threadIdx.x];
    sp[1][threadIdx.x] = P[4096 + m0 + threadIdx.x];
    sp[2][threadIdx.x] = P[8192 + m0 + threadIdx.x];
    __syncthreads();
    for (int j = 0; j < 256; ++j) {
      float dx = qx - sp[0][j], dy = qy - sp[1][j], dz = qz - sp[2][j];
      float d = dx * dx + dy * dy + dz * dz;
      if (d < bd[15]) {
        float cd = d; int ci = m0 + j;
#pragma unroll
        for (int t = 0; t < 16; ++t) {
          if (cd < bd[t]) {
            float td = bd[t]; int ti = bi[t];
            bd[t] = cd; bi[t] = ci; cd = td; ci = ti;
          }
        }
      }
    }
  }
#pragma unroll
  for (int j = 0; j < 16; ++j) idx[((size_t)b * 4096 + n) * 16 + j] = bi[j];
}

// ---------------------------------------------------------------------------
// 1x1-conv GEMM with TDM staging:
//   Y(B,M,4096) = W(MxCk,f16) @ X(B,Ck,4096) + bias.
// Block = 16 columns x all M rows. Wave 0 issues one Tensor-Data-Mover
// descriptor pulling the (Ck x 16) f32 activation tile into LDS (tracked by
// TENSORcnt), all waves then convert once to f16 [n][k] and run WMMA from LDS.
// ---------------------------------------------------------------------------
__global__ __launch_bounds__(256)
void conv1x1_wmma_kernel(const half_t* __restrict__ W, const float* __restrict__ bias,
                         const float* __restrict__ X, float* __restrict__ Y,
                         int M, int Ck) {
  __shared__ float  sXt[256 * 16];   // raw f32 tile  [k][n]   (<= 16 KB)
  __shared__ half_t sXh[16 * 256];   // f16 staging   [n][k]   (<=  8 KB)

  int tid = threadIdx.x, lane = tid & 31, wave = tid >> 5;
  int hi = lane >> 4, ln = lane & 15;
  int g0c = blockIdx.x << 4;          // first column in B*4096 space
  int bb = g0c >> 12;
  int n0 = g0c & 4095;

#if USE_TDM
  if (wave == 0) {
    unsigned ldsOff = (unsigned)(size_t)(&sXt[0]);
    unsigned long long ga =
        (unsigned long long)(size_t)(X + (((size_t)bb * Ck) << 12) + n0);
    u32x4 g0;
    g0[0] = 1u;                                   // count=1 user descriptor
    g0[1] = ldsOff;                               // lds_addr (bytes)
    g0[2] = (unsigned)(ga & 0xffffffffu);         // global_addr[31:0]
    g0[3] = (unsigned)((ga >> 32) & 0x1ffffffu)   // global_addr[56:32]
          | (2u << 30);                           // type = 2 ("image")
    i32x8 g1;
    g1[0] = (int)(2u << 16);                      // data_size = 4 bytes
    g1[1] = (int)(4096u << 16);                   // tensor_dim0[15:0] = 4096
    g1[2] = (int)((unsigned)Ck << 16);            // dim0 hi=0 | tensor_dim1 lo=Ck
    g1[3] = (int)(16u << 16);                     // dim1 hi=0 | tile_dim0 = 16
    g1[4] = (int)(unsigned)Ck;                    // tile_dim1 = Ck, tile_dim2 = 0
    g1[5] = (int)4096u;                           // tensor_dim0_stride[31:0]
    g1[6] = 0;                                    // stride0 hi, stride1 lo
    g1[7] = 0;
    i32x4 gz4 = {0, 0, 0, 0};
    i32x8 gz8 = {0, 0, 0, 0, 0, 0, 0, 0};
    // clang-23 / amdgpu-toolchain form: 6 args
    __builtin_amdgcn_tensor_load_to_lds(g0, g1, gz4, gz4, gz8, 0);
    __builtin_amdgcn_s_wait_tensorcnt(0);
  }
#else
  for (int i = tid; i < Ck * 16; i += 256) {
    int k = i >> 4, n = i & 15;
    sXt[i] = X[(((size_t)bb * Ck + k) << 12) + n0 + n];
  }
#endif
  __syncthreads();

  // transpose+convert once: [k][n] f32 -> [n][k] f16
  for (int i = tid; i < Ck * 16; i += 256) {
    int k = i >> 4, n = i & 15;
    sXh[n * Ck + k] = (half_t)sXt[k * 16 + n];
  }
  __syncthreads();

  int per = (M >> 4) >> 3;            // m-tiles per wave (2 for M=256, 1 for 128)
  for (int i = 0; i < per; ++i) {
    int mt = wave * per + i;
    v8f acc = {};
    for (int k0 = 0; k0 < Ck; k0 += 32) {
      v16h a  = load_a_f16(W, Ck, mt << 4, k0, lane);
      v16h bf = load_b_lds(sXh, Ck, 0, k0, lane);
      acc = wmma32(a, bf, acc);
    }
    int n = n0 + ln;
#pragma unroll
    for (int v = 0; v < 8; ++v) {
      int m = (mt << 4) + (hi << 3) + v;
      Y[(((size_t)bb * M + m) << 12) + n] = acc[v] + bias[m];
    }
  }
}

// ---------------------------------------------------------------------------
// Fused per-neighbor attention kernel (WMMA core).
// ---------------------------------------------------------------------------
__global__ __launch_bounds__(256, 1)
void attn_fused_kernel(const float* __restrict__ pos, const int* __restrict__ idx,
                       const float* __restrict__ key, const float* __restrict__ query,
                       const float* __restrict__ value,
                       const half_t* __restrict__ Wp2h, const float* __restrict__ bp2,
                       const half_t* __restrict__ Wa1h, const float* __restrict__ ba1,
                       const half_t* __restrict__ Wa2h, const float* __restrict__ ba2,
                       const float* __restrict__ Wp1, const float* __restrict__ bp1,
                       const float* __restrict__ sc1, const float* __restrict__ sh1,
                       const float* __restrict__ sc2, const float* __restrict__ sh2,
                       float* __restrict__ agg) {
  __shared__ int    sIdx[32];
  __shared__ float  sQPos[2][3];
  __shared__ float  sW1[64 * 3];
  __shared__ float  sB1[64], sS1[64], sT1[64];
  __shared__ half_t sH1[32 * 64];
  __shared__ half_t sX [32 * 256];
  __shared__ half_t sPE[32 * 256];
  __shared__ half_t sH2[32 * 128];

  int b  = blockIdx.x >> 11;
  int n0 = (blockIdx.x & 2047) << 1;
  int tid = threadIdx.x, lane = tid & 31, wave = tid >> 5;
  int hi = lane >> 4, ln = lane & 15;

  if (tid < 32)  sIdx[tid] = idx[(((size_t)b << 12) + n0 + (tid >> 4)) * 16 + (tid & 15)];
  if (tid < 192) sW1[tid] = Wp1[tid];
  if (tid < 64)  { sB1[tid] = bp1[tid]; sS1[tid] = sc1[tid]; sT1[tid] = sh1[tid]; }
  if (tid < 6)   { int q = tid / 3, d = tid % 3;
                   sQPos[q][d] = pos[(((size_t)b * 3 + d) << 12) + n0 + q]; }
  if (lane == 0) { __builtin_prefetch(Wa1h + (size_t)wave * 32768, 0, 3);
                   __builtin_prefetch(Wa2h + (size_t)wave * 32768, 0, 3); }
  __syncthreads();

  // ---- Phase 1: pos_mlp hidden (K=3, VALU) ----
  {
    int row = tid >> 3;
    int hb  = (tid & 7) << 3;
    int m = sIdx[row], q = row >> 4;
    float p0 = sQPos[q][0] - pos[(((size_t)b * 3 + 0) << 12) + m];
    float p1 = sQPos[q][1] - pos[(((size_t)b * 3 + 1) << 12) + m];
    float p2 = sQPos[q][2] - pos[(((size_t)b * 3 + 2) << 12) + m];
#pragma unroll
    for (int j = 0; j < 8; ++j) {
      int h = hb + j;
      float a = sW1[h * 3] * p0 + sW1[h * 3 + 1] * p1 + sW1[h * 3 + 2] * p2 + sB1[h];
      a = a * sS1[h] + sT1[h];
      sH1[row * 64 + h] = (half_t)fmaxf(a, 0.f);
    }
  }
  __syncthreads();

  // ---- Phase 2: pe = Wp2 @ H1 (WMMA, K=64), x = q - k_gather + pe ----
#pragma unroll
  for (int i = 0; i < 4; ++i) {
    int t = wave * 4 + i;
    int ct = t >> 1, rt = t & 1;
    v8f acc = {};
#pragma unroll
    for (int ks = 0; ks < 2; ++ks) {
      v16h a  = load_a_f16(Wp2h, 64, ct << 4, ks << 5, lane);
      v16h bf = load_b_lds(sH1, 64, rt << 4, ks << 5, lane);
      acc = wmma32(a, bf, acc);
    }
    int row = (rt << 4) + ln;
    int n = n0 + rt;
    int m = sIdx[row];
    int cb = (ct << 4) + (hi << 3);
#pragma unroll
    for (int v = 0; v < 8; ++v) {
      int c = cb + v;
      float pe = acc[v] + bp2[c];
      float x = query[(((size_t)b * 256 + c) << 12) + n]
              - key  [(((size_t)b * 256 + c) << 12) + m] + pe;
      sPE[row * 256 + c] = (half_t)pe;
      sX [row * 256 + c] = (half_t)x;
    }
  }
  __syncthreads();

  // ---- Phase 3: attn MLP 256 -> 1024 -> 256, AH chunked by 128 ----
  v8f acc3[4] = {};
  for (int cc = 0; cc < 8; ++cc) {
#pragma unroll
    for (int i = 0; i < 2; ++i) {            // GEMM2: hidden chunk (bn+relu)
      int t = wave * 2 + i;
      int ht = t >> 1, rt = t & 1;
      int hg = (cc << 7) + (ht << 4);
      v8f a2 = {};
#pragma unroll
      for (int ks = 0; ks < 8; ++ks) {
        v16h a  = load_a_f16(Wa1h, 256, hg, ks << 5, lane);
        v16h bf = load_b_lds(sX, 256, rt << 4, ks << 5, lane);
        a2 = wmma32(a, bf, a2);
      }
      int row = (rt << 4) + ln;
      int hb = hg + (hi << 3);
#pragma unroll
      for (int v = 0; v < 8; ++v) {
        int h = hb + v;
        float tv = (a2[v] + ba1[h]) * sc2[h] + sh2[h];
        sH2[row * 128 + (h - (cc << 7))] = (half_t)fmaxf(tv, 0.f);
      }
    }
    __syncthreads();
#pragma unroll
    for (int i = 0; i < 4; ++i) {            // GEMM3: accumulate logits
      int t = wave * 4 + i;
      int ct = t >> 1, rt = t & 1;
#pragma unroll
      for (int ks = 0; ks < 4; ++ks) {
        v16h a  = load_a_f16(Wa2h, 1024, ct << 4, (cc << 7) + (ks << 5), lane);
        v16h bf = load_b_lds(sH2, 128, rt << 4, ks << 5, lane);
        acc3[i] = wmma32(a, bf, acc3[i]);
      }
    }
    __syncthreads();
  }

  // ---- Phase 4: softmax over K (16 lanes) + weighted aggregation ----
#pragma unroll
  for (int i = 0; i < 4; ++i) {
    int t = wave * 4 + i;
    int ct = t >> 1, rt = t & 1;
    int row = (rt << 4) + ln;
    int m = sIdx[row];
    int n = n0 + rt;
    int cb = (ct << 4) + (hi << 3);
#pragma unroll
    for (int v = 0; v < 8; ++v) {
      int c = cb + v;
      float logit = acc3[i][v] + ba2[c];
      float mx = logit;
      for (int s = 8; s; s >>= 1) mx = fmaxf(mx, __shfl_xor(mx, s, 16));
      float e = __expf(logit - mx);
      float sm = e;
      for (int s = 8; s; s >>= 1) sm += __shfl_xor(sm, s, 16);
      float aw = e / sm;
      float val = value[(((size_t)b * 256 + c) << 12) + m]
                + (float)sPE[row * 256 + c];
      float p = aw * val;
      for (int s = 8; s; s >>= 1) p += __shfl_xor(p, s, 16);
      if (ln == 0) agg[(((size_t)b * 256 + c) << 12) + n] = p;
    }
  }
}

// ---------------------------------------------------------------------------
extern "C" void kernel_launch(void* const* d_in, const int* in_sizes, int n_in,
                              void* d_out, int out_size, void* d_ws, size_t ws_size,
                              hipStream_t stream) {
  (void)in_sizes; (void)n_in; (void)out_size; (void)ws_size;
  const float* obj   = (const float*)d_in[0];
  const float* pos   = (const float*)d_in[1];
  const float* qp    = (const float*)d_in[2];
  const float* W_lsq = (const float*)d_in[3];
  const float* b_lsq = (const float*)d_in[4];
  const float* W_lso = (const float*)d_in[5];
  const float* b_lso = (const float*)d_in[6];
  const float* Wk = (const float*)d_in[7];   const float* bk = (const float*)d_in[8];
  const float* Wq = (const float*)d_in[9];   const float* bq = (const float*)d_in[10];
  const float* Wv = (const float*)d_in[11];  const float* bv = (const float*)d_in[12];
  const float* Wp1 = (const float*)d_in[13]; const float* bp1 = (const float*)d_in[14];
  const float* g1 = (const float*)d_in[15];  const float* bt1 = (const float*)d_in[16];
  const float* m1 = (const float*)d_in[17];  const float* v1  = (const float*)d_in[18];
  const float* Wp2 = (const float*)d_in[19]; const float* bp2 = (const float*)d_in[20];
  const float* Wa1 = (const float*)d_in[21]; const float* ba1 = (const float*)d_in[22];
  const float* g2 = (const float*)d_in[23];  const float* bt2 = (const float*)d_in[24];
  const float* m2 = (const float*)d_in[25];  const float* v2  = (const float*)d_in[26];
  const float* Wa2 = (const float*)d_in[27]; const float* ba2 = (const float*)d_in[28];
  const float* We  = (const float*)d_in[29]; const float* bfin = (const float*)d_in[30];

  char* ws = (char*)d_ws;
  size_t off = 0;
  auto carve = [&](size_t bytes) -> void* {
    off = (off + 255) & ~(size_t)255;
    void* p = ws + off;
    off += bytes;
    return p;
  };
  half_t* Wkh = (half_t*)carve(256 * 128 * 2);
  half_t* Wqh = (half_t*)carve(256 * 128 * 2);
  half_t* Wvh = (half_t*)carve(256 * 128 * 2);
  float*  bkp = (float*)carve(256 * 4);
  float*  bqp = (float*)carve(256 * 4);
  float*  bvp = (float*)carve(256 * 4);
  half_t* Wp2h = (half_t*)carve(256 * 64 * 2);
  half_t* Wa1h = (half_t*)carve(1024 * 256 * 2);
  half_t* Wa2h = (half_t*)carve(256 * 1024 * 2);
  half_t* Weh  = (half_t*)carve(128 * 256 * 2);
  float* sc1b = (float*)carve(64 * 4);
  float* sh1b = (float*)carve(64 * 4);
  float* sc2b = (float*)carve(1024 * 4);
  float* sh2b = (float*)carve(1024 * 4);
  int*   idxb = (int*)carve((size_t)2 * 4096 * 16 * 4);
  float* keyb   = (float*)carve((size_t)2 * 256 * 4096 * 4);
  float* queryb = (float*)carve((size_t)2 * 256 * 4096 * 4);
  float* valueb = (float*)carve((size_t)2 * 256 * 4096 * 4);
  float* aggb   = (float*)carve((size_t)2 * 256 * 4096 * 4);

  fuse_qkv_kernel<<<768, 128, 0, stream>>>(Wk, bk, Wq, bq, Wv, bv,
                                           W_lso, b_lso, W_lsq, b_lsq,
                                           Wkh, bkp, Wqh, bqp, Wvh, bvp);
  f16_convert_kernel<<<64, 256, 0, stream>>>(Wp2h, Wp2, 256 * 64);
  f16_convert_kernel<<<1024, 256, 0, stream>>>(Wa1h, Wa1, 1024 * 256);
  f16_convert_kernel<<<1024, 256, 0, stream>>>(Wa2h, Wa2, 256 * 1024);
  f16_convert_kernel<<<128, 256, 0, stream>>>(Weh, We, 128 * 256);
  bn_fold_kernel<<<1, 64, 0, stream>>>(sc1b, sh1b, g1, bt1, m1, v1, 64);
  bn_fold_kernel<<<4, 256, 0, stream>>>(sc2b, sh2b, g2, bt2, m2, v2, 1024);

  knn_kernel<<<32, 256, 0, stream>>>(pos, idxb);

  // 512 blocks = (B*4096)/16 column groups; TDM-staged WMMA GEMMs
  conv1x1_wmma_kernel<<<512, 256, 0, stream>>>(Wkh, bkp, obj, keyb, 256, 128);
  conv1x1_wmma_kernel<<<512, 256, 0, stream>>>(Wvh, bvp, obj, valueb, 256, 128);
  conv1x1_wmma_kernel<<<512, 256, 0, stream>>>(Wqh, bqp, qp, queryb, 256, 128);

  attn_fused_kernel<<<4096, 256, 0, stream>>>(pos, idxb, keyb, queryb, valueb,
                                              Wp2h, bp2, Wa1h, ba1, Wa2h, ba2,
                                              Wp1, bp1, sc1b, sh1b, sc2b, sh2b,
                                              aggb);

  conv1x1_wmma_kernel<<<512, 256, 0, stream>>>(Weh, bfin, aggb, (float*)d_out,
                                               128, 256);
}
